// Conv2d_mvm_18279380812497
// MI455X (gfx1250) — 8-bit matrix operand order repaired and checked by reading, compiled, not run on hardware
//
#include <hip/hip_runtime.h>
#include <hip/hip_bf16.h>
#include <stdint.h>

typedef int v2i __attribute__((ext_vector_type(2)));
typedef int v4i __attribute__((ext_vector_type(4)));
typedef int v8i __attribute__((ext_vector_type(8)));

#define USE_ASYNC 1

__device__ __forceinline__ int quant_q(float v) {
  float r = rintf(v * 4096.0f);
  r = fminf(fmaxf(r, -32768.0f), 32767.0f);
  return (int)r;
}

__global__ void quant_w_kernel(const float* __restrict__ w,
                               int8_t* __restrict__ whi,
                               uint8_t* __restrict__ wlo) {
  int idx = blockIdx.x * 256 + threadIdx.x;
  if (idx >= 256 * 1152) return;
  int s = idx % 3;
  int t = idx / 3;
  int r = t % 3;  t /= 3;
  int c = t % 128;
  int o = t / 128;
  int q = quant_q(w[idx]);
  int ko = o * 1152 + (r * 3 + s) * 128 + c;
  whi[ko] = (int8_t)(q >> 8);
  wlo[ko] = (uint8_t)(q & 255);
}

__global__ void quant_x_kernel(const float* __restrict__ x,
                               int8_t* __restrict__ xhi,
                               uint8_t* __restrict__ xlo) {
  long idx = (long)blockIdx.x * 256 + threadIdx.x;
  if (idx >= 32l * 58 * 58 * 128) return;
  int c = (int)(idx & 127);
  long t = idx >> 7;
  int pw = (int)(t % 58); t /= 58;
  int ph = (int)(t % 58);
  int b  = (int)(t / 58);
  int q = 0;
  if (ph >= 1 && ph < 57 && pw >= 1 && pw < 57) {
    q = quant_q(x[(((long)b * 128 + c) * 56 + (ph - 1)) * 56 + (pw - 1)]);
  }
  xhi[idx] = (int8_t)(q >> 8);
  xlo[idx] = (uint8_t)(q & 255);
}

__device__ __forceinline__ void stage16(void* lds_dst, const void* gsrc) {
#if USE_ASYNC
  unsigned loff = (unsigned)(uintptr_t)lds_dst;
  asm volatile("global_load_async_to_lds_b128 %0, %1, off"
               :: "v"(loff), "v"(gsrc) : "memory");
#else
  *(v4i*)lds_dst = *(const v4i*)gsrc;
#endif
}
__device__ __forceinline__ void wait_async_le6() {
#if USE_ASYNC
  asm volatile("s_wait_asynccnt 0x6" ::: "memory");
#endif
}
__device__ __forceinline__ void wait_async_le0() {
#if USE_ASYNC
  asm volatile("s_wait_asynccnt 0x0" ::: "memory");
#endif
}

__device__ __forceinline__ v8i load_a_frag(const char* p) {
  v8i a;
#pragma unroll
  for (int j = 0; j < 4; ++j) {
    v2i v = *(const v2i*)(p + (j >> 1) * 32 + (j & 1) * 8);
    a[2 * j]     = v[0];
    a[2 * j + 1] = v[1];
  }
  return a;
}
__device__ __forceinline__ v8i load_b_frag(const char* p) {
  v4i x0 = *(const v4i*)(p);
  v4i x1 = *(const v4i*)(p + 32);
  v8i b;
  b[0] = x0[0]; b[1] = x0[1]; b[2] = x0[2]; b[3] = x0[3];
  b[4] = x1[0]; b[5] = x1[1]; b[6] = x1[2]; b[7] = x1[3];
  return b;
}

__global__ __launch_bounds__(256, 1) void conv_wmma_kernel(
    const int8_t* __restrict__ xhi, const uint8_t* __restrict__ xlo,
    const int8_t* __restrict__ whi, const uint8_t* __restrict__ wlo,
    float* __restrict__ out) {
  constexpr int KSTEPS = 18;
  constexpr int A_SL   = 128 * 80;
  constexpr int B_SL   = 64 * 80;
  constexpr int BUFB   = 2 * A_SL + 2 * B_SL;
  __shared__ __align__(16) char lds[2 * BUFB];

  const int t    = threadIdx.x;
  const int lane = t & 31;
  const int lm   = lane & 15;
  const int hiL  = lane >> 4;
  const int wave = t >> 5;
  const int wm   = wave >> 1;
  const int wn   = wave & 1;
  const int p0       = blockIdx.x * 64;
  const int coutBase = blockIdx.y * 128;

  const int bi  = t >> 2;
  const int bch = (t & 3) << 4;
  const int bp  = p0 + bi;
  const int bb  = bp / 3136;
  const int brem = bp - bb * 3136;
  const int boh = brem / 56;
  const int bow = brem - boh * 56;

  auto stage = [&](int st, int buf) {
    const int rs = st >> 1;
    const int c0 = (st & 1) << 6;
    const int r  = rs / 3;
    const int s  = rs - r * 3;
    char* base = lds + buf * BUFB;
#pragma unroll
    for (int j = 0; j < 2; ++j) {
      int idx = t + j * 256;
      int row = idx >> 2;
      int ch  = (idx & 3) << 4;
      int src = (coutBase + row) * 1152 + rs * 128 + c0 + ch;
      int dst = row * 80 + ch;
      stage16(base + dst,        whi + src);
      stage16(base + A_SL + dst, wlo + src);
    }
    int xs = ((bb * 58 + boh + r) * 58 + (bow + s)) * 128 + c0 + bch;
    int dst = bi * 80 + bch;
    stage16(base + 2 * A_SL + dst,        xhi + xs);
    stage16(base + 2 * A_SL + B_SL + dst, xlo + xs);
  };

  v8i acc_hh[2][2], acc_mid[2][2], acc_ll[2][2];
  const v8i vz = {0, 0, 0, 0, 0, 0, 0, 0};
#pragma unroll
  for (int i = 0; i < 2; ++i)
#pragma unroll
    for (int j = 0; j < 2; ++j) {
      acc_hh[i][j] = vz; acc_mid[i][j] = vz; acc_ll[i][j] = vz;
    }

  stage(0, 0);
  stage(1, 1);

  for (int st = 0; st < KSTEPS; ++st) {
    const int buf = st & 1;
    if (st < KSTEPS - 1) wait_async_le6();
    else                 wait_async_le0();
    __syncthreads();

    const char* base = lds + buf * BUFB;
    const char* Ah = base;
    const char* Bh = base + 2 * A_SL;

    v8i ah[2], al[2], bh[2], bl[2];
#pragma unroll
    for (int mt = 0; mt < 2; ++mt) {
      const char* pA = Ah + (wm * 32 + mt * 16 + lm) * 80 + hiL * 16;
      ah[mt] = load_a_frag(pA);
      al[mt] = load_a_frag(pA + A_SL);
    }
#pragma unroll
    for (int nt = 0; nt < 2; ++nt) {
      const char* pB = Bh + (wn * 32 + nt * 16 + lm) * 80 + hiL * 16;
      bh[nt] = load_b_frag(pB);
      bl[nt] = load_b_frag(pB + B_SL);
    }

#pragma unroll
    for (int mt = 0; mt < 2; ++mt)
#pragma unroll
      for (int nt = 0; nt < 2; ++nt) {
        acc_hh[mt][nt] = __builtin_amdgcn_wmma_i32_16x16x64_iu8(
            true,  ah[mt], true,  bh[nt], acc_hh[mt][nt], false, false);
        acc_mid[mt][nt] = __builtin_amdgcn_wmma_i32_16x16x64_iu8(
            true,  ah[mt], false, bl[nt], acc_mid[mt][nt], false, false);
        acc_mid[mt][nt] = __builtin_amdgcn_wmma_i32_16x16x64_iu8(
            false, al[mt], true,  bh[nt], acc_mid[mt][nt], false, false);
        acc_ll[mt][nt] = __builtin_amdgcn_wmma_i32_16x16x64_iu8(
            false, al[mt], false, bl[nt], acc_ll[mt][nt], false, false);
      }

    __syncthreads();
    if (st + 2 < KSTEPS) stage(st + 2, buf);
  }

#pragma unroll
  for (int mt = 0; mt < 2; ++mt) {
    const int cb = coutBase + wm * 32 + mt * 16 + hiL * 8;
#pragma unroll
    for (int nt = 0; nt < 2; ++nt) {
      const int pix  = p0 + wn * 32 + nt * 16 + lm;
      const int ob   = pix / 3136;
      const int orem = pix - ob * 3136;
      const int ooh  = orem / 56;
      const int oow  = orem - ooh * 56;
      float* op = out + (((long)ob * 256 + cb) * 56 + ooh) * 56 + oow;
#pragma unroll
      for (int rr = 0; rr < 8; ++rr) {
        long q = ((long)acc_hh[mt][nt][rr] << 16) +
                 ((long)acc_mid[mt][nt][rr] << 8) +
                 (long)acc_ll[mt][nt][rr];
        op[(long)rr * 3136] = (float)q * (1.0f / 16777216.0f);
      }
    }
  }
}

extern "C" void kernel_launch(void* const* d_in, const int* in_sizes, int n_in,
                              void* d_out, int out_size, void* d_ws, size_t ws_size,
                              hipStream_t stream) {
  const float* x = (const float*)d_in[0];
  const float* w = (const float*)d_in[1];
  float* out = (float*)d_out;

  char* ws = (char*)d_ws;
  const size_t XP = 32ull * 58 * 58 * 128;
  const size_t WP = 256ull * 1152;
  int8_t*  xhi = (int8_t*)ws;
  uint8_t* xlo = (uint8_t*)(ws + XP);
  int8_t*  whi = (int8_t*)(ws + 2 * XP);
  uint8_t* wlo = (uint8_t*)(ws + 2 * XP + WP);

  quant_x_kernel<<<(unsigned)((XP + 255) / 256), 256, 0, stream>>>(x, xhi, xlo);
  quant_w_kernel<<<(unsigned)((WP + 255) / 256), 256, 0, stream>>>(w, whi, wlo);

  conv_wmma_kernel<<<dim3(1568, 2, 1), 256, 0, stream>>>(xhi, xlo, whi, wlo, out);
}
